// MLP_48687749268221
// MI455X (gfx1250) — matrix operand order repaired and checked by reading, not compiled, not run on hardware
//
#include <hip/hip_runtime.h>
#include <hip/hip_bf16.h>
#include <stdint.h>

typedef __bf16 bf16_t;
typedef __attribute__((ext_vector_type(16))) bf16_t v16bf;
typedef __attribute__((ext_vector_type(8)))  bf16_t v8bf;
typedef __attribute__((ext_vector_type(8)))  float  v8f;

#define IN_UNITS 128
#define K1 256          // 2*IN_UNITS
#define H1 128
#define H2 64
#define WPB 8           // waves per block
#define THREADS (WPB*32)
#define H1STRIDE 136    // padded row stride (bf16) -> conflict-free h1 scatter
#define H1TILE   (16*H1STRIDE)

// Build a WMMA A-layout 16-element fragment from two 8-element runs (f32 src).
__device__ __forceinline__ v16bf make_frag_f32(const float* src) {
    v8f r0 = *(const v8f*)(src);
    v8f r1 = *(const v8f*)(src + 16);
    v8bf lo = __builtin_convertvector(r0, v8bf);   // packed v_cvt_pk_bf16_f32
    v8bf hi = __builtin_convertvector(r1, v8bf);
    return __builtin_shufflevector(lo, hi, 0,1,2,3,4,5,6,7,8,9,10,11,12,13,14,15);
}

__device__ __forceinline__ v16bf make_frag_bf16(const bf16_t* src) {
    v8bf r0 = *(const v8bf*)(src);
    v8bf r1 = *(const v8bf*)(src + 16);
    return __builtin_shufflevector(r0, r1, 0,1,2,3,4,5,6,7,8,9,10,11,12,13,14,15);
}

__device__ __forceinline__ v8f splat8(float x) {
    v8f s = { x, x, x, x, x, x, x, x };
    return s;
}

__global__ __launch_bounds__(THREADS)
void mlp_edge_kernel(const float* __restrict__ drug,   // [10000,128]
                     const float* __restrict__ dis,    // [5000,128]
                     const int*   __restrict__ eidx,   // [2,E]
                     const float* __restrict__ W1,     // [128,256]
                     const float* __restrict__ B1,     // [128]
                     const float* __restrict__ W2,     // [64,128]
                     const float* __restrict__ B2,     // [64]
                     const float* __restrict__ W3,     // [64]
                     const float* __restrict__ B3,     // [1]
                     float* __restrict__ out,          // [E]
                     int E)
{
    // Weight fragments in WMMA B-layout: frag f = (nt,kt); per lane 16 bf16:
    //   N = nt*16 + (lane&15),  K = kt*32 + 8*(lane>>4) + (e&7) + 16*(e>>3)
    __shared__ __attribute__((aligned(32))) bf16_t sW1[64*512];       // 64 KB
    __shared__ __attribute__((aligned(32))) bf16_t sW2[16*512];       // 16 KB
    __shared__ float  sB1[128];
    __shared__ float  sB2[64];
    __shared__ float  sW3[64];
    // Per-wave scratch: h1 for two M-tiles, padded rows (2*16*136 bf16 ~ 8.5 KB);
    // reused as the layer-3 partial buffer after layer-2 A-fragments are read.
    __shared__ __attribute__((aligned(32))) bf16_t sH1[WPB][2*H1TILE]; // ~68 KB

    const int tid   = threadIdx.x;
    const int lane  = tid & 31;
    const int wave  = tid >> 5;
    const int lrow  = lane & 15;   // M row / N col within a 16-tile
    const int lhalf = lane >> 4;

    // ---- one-time weight staging (block-wide) ----
    for (int i = tid; i < 64*512; i += THREADS) {
        int f = i >> 9, w = i & 511;
        int l = w >> 4, e = w & 15;
        int nt = f >> 3, kt = f & 7;
        int n = nt*16 + (l & 15);
        int k = kt*32 + 8*(l >> 4) + (e & 7) + 16*(e >> 3);
        sW1[i] = (bf16_t)W1[n*K1 + k];
    }
    for (int i = tid; i < 16*512; i += THREADS) {
        int f = i >> 9, w = i & 511;
        int l = w >> 4, e = w & 15;
        int nt = f >> 2, kt = f & 3;
        int n = nt*16 + (l & 15);
        int k = kt*32 + 8*(l >> 4) + (e & 7) + 16*(e >> 3);
        sW2[i] = (bf16_t)W2[n*H1 + k];
    }
    if (tid < 128) sB1[tid] = B1[tid];
    if (tid < 64)  { sB2[tid] = B2[tid]; sW3[tid] = W3[tid]; }
    __syncthreads();

    const float b3v  = B3[0];
    const int npair  = (E + 31) >> 5;            // 32 edges per wave-iteration
    const int gwave  = blockIdx.x * WPB + wave;
    const int wstep  = gridDim.x * WPB;
    bf16_t* h1w = sH1[wave];

    for (int t = gwave; t < npair; t += wstep) {
        const int ebase = t << 5;
        const int erA = min(ebase + lrow,      E - 1);
        const int erB = min(ebase + 16 + lrow, E - 1);
        const float* drowA = drug + (long)eidx[erA]     * IN_UNITS;
        const float* srowA = dis  + (long)eidx[E + erA] * IN_UNITS;
        const float* drowB = drug + (long)eidx[erB]     * IN_UNITS;
        const float* srowB = dis  + (long)eidx[E + erB] * IN_UNITS;

        // ================= layer 1: two 16x256x128 GEMMs =================
        // kt-outer / nt-inner, 16 persistent accumulators, B double-buffered.
        v8f acc1[16];
        #pragma unroll
        for (int nt = 0; nt < 8; ++nt) {
            v8f s = splat8(sB1[nt*16 + lrow]);
            acc1[2*nt]   = s;
            acc1[2*nt+1] = s;
        }
        #pragma unroll
        for (int kt = 0; kt < 8; ++kt) {
            int base = kt*32 + 8*lhalf;
            const float* sA = (base < 128) ? (drowA + base) : (srowA + (base - 128));
            const float* sB = (base < 128) ? (drowB + base) : (srowB + (base - 128));
            v16bf aA = make_frag_f32(sA);
            v16bf aB = make_frag_f32(sB);
            v16bf bcur = *(const v16bf*)(sW1 + kt*512 + lane*16);   // nt = 0
            #pragma unroll
            for (int nt = 0; nt < 8; ++nt) {
                v16bf bnext = bcur;
                if (nt < 7)  // static under full unroll: prefetch next B
                    bnext = *(const v16bf*)(sW1 + ((nt+1)*8 + kt)*512 + lane*16);
                acc1[2*nt]   = __builtin_amdgcn_wmma_f32_16x16x32_bf16(
                                   false, aA, false, bcur, (short)0, acc1[2*nt],   false, false);
                acc1[2*nt+1] = __builtin_amdgcn_wmma_f32_16x16x32_bf16(
                                   false, aB, false, bcur, (short)0, acc1[2*nt+1], false, false);
                bcur = bnext;
            }
        }
        // relu -> bf16 -> per-wave LDS, padded row-major [16][H1STRIDE] per tile
        #pragma unroll
        for (int nt = 0; nt < 8; ++nt) {
            int col = nt*16 + lrow;
            v8f rA, rB;
            #pragma unroll
            for (int r = 0; r < 8; ++r) { rA[r] = fmaxf(acc1[2*nt][r],   0.0f);
                                          rB[r] = fmaxf(acc1[2*nt+1][r], 0.0f); }
            v8bf hA = __builtin_convertvector(rA, v8bf);
            v8bf hB = __builtin_convertvector(rB, v8bf);
            #pragma unroll
            for (int r = 0; r < 8; ++r) {
                int m = r + 8*lhalf;               // C/D layout: M = r + 8*(lane>>4)
                h1w[m*H1STRIDE + col]          = hA[r];
                h1w[H1TILE + m*H1STRIDE + col] = hB[r];
            }
        }
        asm volatile("" ::: "memory");

        // ================= layer 2: two 16x128x64 GEMMs =================
        v8f acc2[8];
        #pragma unroll
        for (int nt = 0; nt < 4; ++nt) {
            v8f s = splat8(sB2[nt*16 + lrow]);
            acc2[2*nt]   = s;
            acc2[2*nt+1] = s;
        }
        #pragma unroll
        for (int kt = 0; kt < 4; ++kt) {
            int base = kt*32 + 8*lhalf;
            v16bf aA = make_frag_bf16(h1w + lrow*H1STRIDE + base);
            v16bf aB = make_frag_bf16(h1w + H1TILE + lrow*H1STRIDE + base);
            v16bf bcur = *(const v16bf*)(sW2 + kt*512 + lane*16);   // nt = 0
            #pragma unroll
            for (int nt = 0; nt < 4; ++nt) {
                v16bf bnext = bcur;
                if (nt < 3)
                    bnext = *(const v16bf*)(sW2 + ((nt+1)*4 + kt)*512 + lane*16);
                acc2[2*nt]   = __builtin_amdgcn_wmma_f32_16x16x32_bf16(
                                   false, aA, false, bcur, (short)0, acc2[2*nt],   false, false);
                acc2[2*nt+1] = __builtin_amdgcn_wmma_f32_16x16x32_bf16(
                                   false, aB, false, bcur, (short)0, acc2[2*nt+1], false, false);
                bcur = bnext;
            }
        }
        asm volatile("" ::: "memory");

        // ---- layer 3: per-lane partial dot with W3 ----
        float pA[8] = {0,0,0,0,0,0,0,0};
        float pB[8] = {0,0,0,0,0,0,0,0};
        #pragma unroll
        for (int nt = 0; nt < 4; ++nt) {
            float w3n = sW3[nt*16 + lrow];
            #pragma unroll
            for (int r = 0; r < 8; ++r) {
                pA[r] += fmaxf(acc2[2*nt][r],   0.0f) * w3n;
                pB[r] += fmaxf(acc2[2*nt+1][r], 0.0f) * w3n;
            }
        }

        // ---- 16-lane reductions through per-wave LDS (aliases h1) ----
        float* pw = (float*)h1w;                   // h1 is dead now; reuse region
        #pragma unroll
        for (int r = 0; r < 8; ++r) {
            pw[lane*8 + r]       = pA[r];
            pw[256 + lane*8 + r] = pB[r];
        }
        asm volatile("" ::: "memory");
        {
            int tile = lane >> 4;                  // 0 = tileA, 1 = tileB
            int mm   = lane & 15;                  // edge row within tile
            int sb   = (mm & 8) ? 16 : 0;          // M=r from lanes 0-15, M=r+8 from 16-31
            int r    = mm & 7;
            const float* q = pw + tile*256 + sb*8 + r;
            float s = b3v;
            #pragma unroll
            for (int src = 0; src < 16; ++src) s += q[src*8];
            int e = ebase + lane;                  // tile*16 + mm == lane
            if (e < E) out[e] = s;
        }
        asm volatile("" ::: "memory");
    }
}

extern "C" void kernel_launch(void* const* d_in, const int* in_sizes, int n_in,
                              void* d_out, int out_size, void* d_ws, size_t ws_size,
                              hipStream_t stream) {
    const float* drug = (const float*)d_in[0];
    const float* dis  = (const float*)d_in[1];
    const int*   eidx = (const int*)d_in[2];
    const float* W1   = (const float*)d_in[3];
    const float* B1   = (const float*)d_in[4];
    const float* W2   = (const float*)d_in[5];
    const float* B2   = (const float*)d_in[6];
    const float* W3   = (const float*)d_in[7];
    const float* B3   = (const float*)d_in[8];
    float* out = (float*)d_out;

    int E = in_sizes[2] / 2;            // edge_index is [2, E]
    int npair = (E + 31) / 32;          // 32 edges per wave-iteration
    int blocks = (npair + WPB - 1) / WPB;
    if (blocks > 2048) blocks = 2048;   // grid-stride; amortizes LDS weight staging

    mlp_edge_kernel<<<blocks, THREADS, 0, stream>>>(
        drug, dis, eidx, W1, B1, W2, B2, W3, B3, out, E);
}
